// EMGMambaRegressor_52089363366469
// MI455X (gfx1250) — compile-verified, matrix operand order checked by reading, not run on hardware
//
#include <hip/hip_runtime.h>
#include <hip/hip_bf16.h>

// ---------------------------------------------------------------------------
// EMG Mamba regressor for MI455X (gfx1250, wave32, WMMA).
//
// Shapes: B=8, T=1024 (M = B*T = 8192 rows), D_MODEL=256, D_INNER=512,
// D_STATE=16, DT_RANK=16, D_CONV=4, DEPTH=5, OUT=10.
//
// All four projections per layer are out[m,n] = sum_k act[m,k] * W[n,k]
// (W row-major in k), so the WMMA B-operand (K x N, column n per lane) is
// loaded from W rows exactly like the A-operand (M x K, row m per lane):
// contiguous-K 16B vector loads, no transpose needed. One wave computes an
// NT-wide strip of 16x16 C tiles (A tile reused NT times). The K-loop is
// software-pipelined (double-buffered); __launch_bounds__(256, 1) raises the
// VGPR budget so prefetched tiles keep distinct registers and loads stay in
// flight across the v_wmma_f32_16x16x32_f16 ops. K is a template parameter
// so no tail/guard code exists; the K=16 dt-projection is handled by
// zero-padding operands to K=32.
//
// 16-bit A/B tile layout per CDNA5 ISA (CDNA5 ISA): lane<16 holds
// K {kk+0..7, kk+16..23} of row (lane&15); lane>=16 holds K {kk+8..15,
// kk+24..31}. C/D: lane gives n=lane&15, VGPR r gives m = (lane>>4)*8 + r.
// ---------------------------------------------------------------------------

#define D_MODEL 256
#define DEPTH   5
#define D_STATE 16
#define D_CONV  4
#define D_INNER 512
#define DT_RANK 16
#define IN_DIM  12
#define OUT_DIM 10
#define BB      8
#define TT      1024
#define MROWS   (BB * TT)   // 8192

typedef __attribute__((ext_vector_type(16))) _Float16 v16h;
typedef __attribute__((ext_vector_type(8)))  float    v8f;
typedef __attribute__((ext_vector_type(8)))  _Float16 v8h;

__device__ __forceinline__ float silu_f(float x) {
  return x / (1.0f + __expf(-x));
}

__device__ __forceinline__ float softplus_f(float x) {
  return (x > 20.0f) ? x : log1pf(__expf(x));
}

// ---------------- f32 -> f16 cast (weights / activations) ------------------
__global__ void cast_f16_kernel(const float* __restrict__ in,
                                _Float16* __restrict__ out, int n) {
  int idx = blockIdx.x * 256 + threadIdx.x;
  if (idx < n) out[idx] = (_Float16)in[idx];
}

// f32 [rows, in_ld] -> f16 [rows, out_ld], cols >= cols_copy zero-filled
__global__ void pad_cast_f16_kernel(const float* __restrict__ in, int in_ld,
                                    _Float16* __restrict__ out, int out_ld,
                                    int rows, int cols_copy) {
  int idx = blockIdx.x * 256 + threadIdx.x;
  if (idx >= rows * out_ld) return;
  int r = idx / out_ld, c = idx - r * out_ld;
  out[idx] = (c < cols_copy) ? (_Float16)in[(size_t)r * in_ld + c]
                             : (_Float16)0.0f;
}

// ---------------- WMMA tile load: 16x32 f16, wave32 layout -----------------
// p already points at row (lane&15), K-offset (lane>=16 ? 8 : 0).
__device__ __forceinline__ v16h load_tile16x32(const _Float16* __restrict__ p) {
  v8h lo = *(const v8h*)(p);        // K = kk + hi      .. +7
  v8h hh = *(const v8h*)(p + 16);   // K = kk + 16 + hi .. +7
  return __builtin_shufflevector(lo, hh, 0, 1, 2, 3, 4, 5, 6, 7,
                                 8, 9, 10, 11, 12, 13, 14, 15);
}

// ---------------- WMMA GEMM: C[M,N] = A[M,K] * W[N,K]^T --------------------
// One wave owns NT consecutive 16x16 tiles along N (A tile reused NT times).
// Double-buffered K pipeline: next tiles load while current WMMAs issue.
// min-waves-per-EU = 1 lets the allocator keep both tile buffers live.
template <int K, int NT>
__global__ __launch_bounds__(256, 1)
void gemm_wmma_kernel(const _Float16* __restrict__ A, int lda,
                      const _Float16* __restrict__ W, int ldb,
                      float* __restrict__ C, int ldc,
                      int M, int N) {
  const int wave     = blockIdx.x * (blockDim.x >> 5) + (threadIdx.x >> 5);
  const int groups_n = (N >> 4) / NT;
  const int total    = (M >> 4) * groups_n;
  if (wave >= total) return;                // wave-uniform: EXEC stays full
  const int m0 = (wave / groups_n) << 4;
  const int n0 = (wave % groups_n) * (16 * NT);

  const int lane = threadIdx.x & 31;
  const int rsub = lane & 15;
  const int hi   = (lane >> 4) << 3;        // 0 or 8

  const _Float16* aptr = A + (size_t)(m0 + rsub) * lda + hi;
  const _Float16* bptr[NT];
#pragma unroll
  for (int nt = 0; nt < NT; ++nt)
    bptr[nt] = W + (size_t)(n0 + nt * 16 + rsub) * ldb + hi;

  v8f acc[NT];
#pragma unroll
  for (int nt = 0; nt < NT; ++nt) acc[nt] = (v8f){};

  // pipeline prologue: first K-step tiles
  v16h a0 = load_tile16x32(aptr);
  v16h b0[NT];
#pragma unroll
  for (int nt = 0; nt < NT; ++nt) b0[nt] = load_tile16x32(bptr[nt]);

#pragma unroll
  for (int kk = 0; kk < K - 32; kk += 32) {
    // issue next K-step loads into fresh registers (overlap with WMMAs below)
    v16h a1 = load_tile16x32(aptr + kk + 32);
    v16h b1[NT];
#pragma unroll
    for (int nt = 0; nt < NT; ++nt) b1[nt] = load_tile16x32(bptr[nt] + kk + 32);
#pragma unroll
    for (int nt = 0; nt < NT; ++nt)
      acc[nt] = __builtin_amdgcn_wmma_f32_16x16x32_f16(
          false, a0, false, b0[nt], (short)0, acc[nt], false, false);
    a0 = a1;
#pragma unroll
    for (int nt = 0; nt < NT; ++nt) b0[nt] = b1[nt];
  }
  // epilogue: last K-step
#pragma unroll
  for (int nt = 0; nt < NT; ++nt)
    acc[nt] = __builtin_amdgcn_wmma_f32_16x16x32_f16(
        false, a0, false, b0[nt], (short)0, acc[nt], false, false);

  const int n_lane = lane & 15;
  const int mb     = m0 + ((lane >> 4) << 3);
#pragma unroll
  for (int nt = 0; nt < NT; ++nt) {
    float* cp = C + (size_t)mb * ldc + n0 + nt * 16 + n_lane;
#pragma unroll
    for (int r = 0; r < 8; ++r) cp[(size_t)r * ldc] = acc[nt][r];
  }
}

// ---------------- stem Conv1d(12 -> 256, k=3, pad=1) -----------------------
__global__ void stem_kernel(const float* __restrict__ x,
                            const float* __restrict__ w,
                            const float* __restrict__ bias,
                            float* __restrict__ h32, _Float16* __restrict__ h16) {
  size_t idx = (size_t)blockIdx.x * 256 + threadIdx.x;   // over MROWS*D_MODEL
  int d = idx & (D_MODEL - 1);
  size_t bt = idx >> 8;
  int b = (int)(bt >> 10), t = (int)(bt & (TT - 1));
  float acc = bias[d];
#pragma unroll
  for (int k = 0; k < 3; ++k) {
    int tt = t + k - 1;
    if (tt >= 0 && tt < TT) {
      const float* xp = x + ((size_t)b * TT + tt) * IN_DIM;
#pragma unroll
      for (int c = 0; c < IN_DIM; ++c) acc += xp[c] * w[d * (IN_DIM * 3) + c * 3 + k];
    }
  }
  h32[idx] = acc;
  h16[idx] = (_Float16)acc;
}

// ---------------- depthwise causal conv (k=4) + SiLU -----------------------
__global__ void dwconv_kernel(const float* __restrict__ xz,
                              const float* __restrict__ cw,
                              const float* __restrict__ cb,
                              float* __restrict__ xd32, _Float16* __restrict__ xd16,
                              int l) {
  size_t idx = (size_t)blockIdx.x * 256 + threadIdx.x;   // over MROWS*D_INNER
  int i = idx & (D_INNER - 1);
  size_t bt = idx >> 9;
  int b = (int)(bt >> 10), t = (int)(bt & (TT - 1));
  float acc = cb[l * D_INNER + i];
#pragma unroll
  for (int k = 0; k < D_CONV; ++k) {
    int tt = t - (D_CONV - 1) + k;
    if (tt >= 0)
      acc += xz[((size_t)b * TT + tt) * (2 * D_INNER) + i] *
             cw[((size_t)l * D_INNER + i) * D_CONV + k];
  }
  float s = silu_f(acc);
  xd32[idx] = s;
  xd16[idx] = (_Float16)s;
}

// ---------------- selective scan: thread per (b, i) channel ----------------
// Bm/Cm (32 floats per (b,t)) staged into LDS once per timestep.
__global__ void scan_kernel(const float* __restrict__ xdbl,   // [M,48]
                            const float* __restrict__ dtraw,  // [M,512]
                            const float* __restrict__ xd,     // [M,512]
                            const float* __restrict__ A_log,
                            const float* __restrict__ dt_b,
                            float* __restrict__ y, int l) {
  int b = blockIdx.x >> 1;                                 // 2 blocks per batch
  int i = ((blockIdx.x & 1) << 8) + threadIdx.x;           // channel
  float A[D_STATE];
#pragma unroll
  for (int s = 0; s < D_STATE; ++s)
    A[s] = -__expf(A_log[((size_t)l * D_INNER + i) * D_STATE + s]);
  float dtb = dt_b[l * D_INNER + i];
  float hs[D_STATE];
#pragma unroll
  for (int s = 0; s < D_STATE; ++s) hs[s] = 0.0f;

  __shared__ float sh[32];                                 // Bm[16] || Cm[16]
  for (int t = 0; t < TT; ++t) {
    size_t bt = (size_t)b * TT + t;
    __syncthreads();
    if (threadIdx.x < 32) sh[threadIdx.x] = xdbl[bt * 48 + DT_RANK + threadIdx.x];
    __syncthreads();
    float dtv = softplus_f(dtraw[bt * D_INNER + i] + dtb);
    float xv  = xd[bt * D_INNER + i];
    float acc = 0.0f;
#pragma unroll
    for (int s = 0; s < D_STATE; ++s) {
      float dA = __expf(dtv * A[s]);
      hs[s] = dA * hs[s] + (dtv * sh[s]) * xv;
      acc  += hs[s] * sh[16 + s];
    }
    y[bt * D_INNER + i] = acc;
  }
}

// ---------------- y_act = (y + xd*Dp) * silu(z) -> f16 ---------------------
__global__ void yact_kernel(const float* __restrict__ y,
                            const float* __restrict__ xd,
                            const float* __restrict__ xz,
                            const float* __restrict__ Dp,
                            _Float16* __restrict__ yact, int l) {
  size_t idx = (size_t)blockIdx.x * 256 + threadIdx.x;     // over MROWS*D_INNER
  int i = idx & (D_INNER - 1);
  size_t bt = idx >> 9;
  float v  = y[idx] + xd[idx] * Dp[l * D_INNER + i];
  float z  = xz[bt * (2 * D_INNER) + D_INNER + i];
  yact[idx] = (_Float16)(v * silu_f(z));
}

// ---------------- residual add + LayerNorm (block per row) -----------------
__global__ void resid_ln_kernel(const float* __restrict__ delta,
                                const float* __restrict__ g,
                                const float* __restrict__ be,
                                float* __restrict__ h32, _Float16* __restrict__ h16) {
  int row = blockIdx.x, d = threadIdx.x;
  size_t idx = (size_t)row * D_MODEL + d;
  float v = h32[idx] + delta[idx];
  __shared__ float red[D_MODEL];
  red[d] = v;
  __syncthreads();
  for (int s = D_MODEL / 2; s > 0; s >>= 1) {
    if (d < s) red[d] += red[d + s];
    __syncthreads();
  }
  float mean = red[0] * (1.0f / D_MODEL);
  __syncthreads();
  float c = v - mean;
  red[d] = c * c;
  __syncthreads();
  for (int s = D_MODEL / 2; s > 0; s >>= 1) {
    if (d < s) red[d] += red[d + s];
    __syncthreads();
  }
  float var = red[0] * (1.0f / D_MODEL);
  float o = c * rsqrtf(var + 1e-5f) * g[d] + be[d];
  h32[idx] = o;
  h16[idx] = (_Float16)o;
}

// ---------------- mean over T, then head linear ----------------------------
__global__ void mean_kernel(const float* __restrict__ h32, float* __restrict__ hmean) {
  int b = blockIdx.x, d = threadIdx.x;
  float s = 0.0f;
  for (int t = 0; t < TT; ++t) s += h32[((size_t)b * TT + t) * D_MODEL + d];
  hmean[b * D_MODEL + d] = s * (1.0f / TT);
}

__global__ void head_kernel(const float* __restrict__ hmean,
                            const float* __restrict__ opw,
                            const float* __restrict__ opb,
                            float* __restrict__ out) {
  int idx = blockIdx.x * 128 + threadIdx.x;
  if (idx >= BB * OUT_DIM) return;
  int b = idx / OUT_DIM, o = idx % OUT_DIM;
  float s = opb[o];
  for (int d = 0; d < D_MODEL; ++d) s += hmean[b * D_MODEL + d] * opw[o * D_MODEL + d];
  out[idx] = s;
}

// ---------------------------------------------------------------------------
extern "C" void kernel_launch(void* const* d_in, const int* in_sizes, int n_in,
                              void* d_out, int out_size, void* d_ws, size_t ws_size,
                              hipStream_t stream) {
  const float* x       = (const float*)d_in[0];
  const float* conv1_w = (const float*)d_in[1];
  const float* conv1_b = (const float*)d_in[2];
  const float* ln_g    = (const float*)d_in[3];
  const float* ln_b    = (const float*)d_in[4];
  const float* in_w    = (const float*)d_in[5];
  const float* conv_w  = (const float*)d_in[6];
  const float* conv_b  = (const float*)d_in[7];
  const float* xproj_w = (const float*)d_in[8];
  const float* dt_w    = (const float*)d_in[9];
  const float* dt_b    = (const float*)d_in[10];
  const float* A_log   = (const float*)d_in[11];
  const float* Dp      = (const float*)d_in[12];
  const float* out_w   = (const float*)d_in[13];
  const float* op_w    = (const float*)d_in[14];
  const float* op_b    = (const float*)d_in[15];
  float* out = (float*)d_out;

  // bump allocator over d_ws (256B aligned)
  char* ws = (char*)d_ws;
  size_t off = 0;
  auto alloc = [&](size_t bytes) -> void* {
    void* p = ws + off;
    off = (off + bytes + 255) & ~(size_t)255;
    return p;
  };

  float*    h32       = (float*)   alloc((size_t)MROWS * D_MODEL * 4);
  _Float16* h16       = (_Float16*)alloc((size_t)MROWS * D_MODEL * 2);
  float*    xz        = (float*)   alloc((size_t)MROWS * 2 * D_INNER * 4);
  float*    xd32      = (float*)   alloc((size_t)MROWS * D_INNER * 4);
  _Float16* xd16      = (_Float16*)alloc((size_t)MROWS * D_INNER * 2);
  float*    xdbl32    = (float*)   alloc((size_t)MROWS * 48 * 4);
  _Float16* xdblpad16 = (_Float16*)alloc((size_t)MROWS * 32 * 2);       // K=32 padded
  float*    dtraw     = (float*)   alloc((size_t)MROWS * D_INNER * 4);
  float*    yv        = (float*)   alloc((size_t)MROWS * D_INNER * 4);
  _Float16* yact16    = (_Float16*)alloc((size_t)MROWS * D_INNER * 2);
  float*    delta     = (float*)   alloc((size_t)MROWS * D_MODEL * 4);
  _Float16* inw16     = (_Float16*)alloc((size_t)DEPTH * 2 * D_INNER * D_MODEL * 2);
  _Float16* xprojw16  = (_Float16*)alloc((size_t)DEPTH * 48 * D_INNER * 2);
  _Float16* dtwpad16  = (_Float16*)alloc((size_t)DEPTH * D_INNER * 32 * 2);  // K=32 padded
  _Float16* outw16    = (_Float16*)alloc((size_t)DEPTH * D_MODEL * D_INNER * 2);
  float*    hmean     = (float*)   alloc((size_t)BB * D_MODEL * 4);
  (void)ws_size; (void)n_in; (void)in_sizes; (void)out_size;

  auto cast = [&](const float* src, _Float16* dst, int n) {
    cast_f16_kernel<<<(n + 255) / 256, 256, 0, stream>>>(src, dst, n);
  };

  // f16 weight staging (deterministic, every call)
  cast(in_w,    inw16,    DEPTH * 2 * D_INNER * D_MODEL);
  cast(xproj_w, xprojw16, DEPTH * 48 * D_INNER);
  cast(out_w,   outw16,   DEPTH * D_MODEL * D_INNER);
  // dt_w: [DEPTH*512, 16] -> f16 [DEPTH*512, 32] zero-padded
  {
    int n = DEPTH * D_INNER * 32;
    pad_cast_f16_kernel<<<(n + 255) / 256, 256, 0, stream>>>(
        dt_w, DT_RANK, dtwpad16, 32, DEPTH * D_INNER, DT_RANK);
  }

  // stem conv -> h (fp32 + f16)
  stem_kernel<<<(MROWS * D_MODEL) / 256, 256, 0, stream>>>(x, conv1_w, conv1_b, h32, h16);

  for (int l = 0; l < DEPTH; ++l) {
    // xz = h @ in_w^T            [8192,1024]  K=256, NT=4
    {
      int waves = (MROWS >> 4) * ((2 * D_INNER >> 4) / 4);
      gemm_wmma_kernel<D_MODEL, 4><<<(waves + 7) / 8, 256, 0, stream>>>(
          h16, D_MODEL, inw16 + (size_t)l * 2 * D_INNER * D_MODEL, D_MODEL,
          xz, 2 * D_INNER, MROWS, 2 * D_INNER);
    }
    // depthwise causal conv + SiLU -> xd
    dwconv_kernel<<<(MROWS * D_INNER) / 256, 256, 0, stream>>>(
        xz, conv_w, conv_b, xd32, xd16, l);
    // x_dbl = xd @ xproj_w^T     [8192,48]   K=512, NT=3
    {
      int waves = (MROWS >> 4) * ((48 >> 4) / 3);
      gemm_wmma_kernel<D_INNER, 3><<<(waves + 7) / 8, 256, 0, stream>>>(
          xd16, D_INNER, xprojw16 + (size_t)l * 48 * D_INNER, D_INNER,
          xdbl32, 48, MROWS, 48);
    }
    // dt slice -> f16 [M,32] zero-padded
    {
      int n = MROWS * 32;
      pad_cast_f16_kernel<<<(n + 255) / 256, 256, 0, stream>>>(
          xdbl32, 48, xdblpad16, 32, MROWS, DT_RANK);
    }
    // dt_raw = x_dbl[:, :16] @ dt_w^T   [8192,512]  K=32 (padded), NT=4
    {
      int waves = (MROWS >> 4) * ((D_INNER >> 4) / 4);
      gemm_wmma_kernel<32, 4><<<(waves + 7) / 8, 256, 0, stream>>>(
          xdblpad16, 32, dtwpad16 + (size_t)l * D_INNER * 32, 32,
          dtraw, D_INNER, MROWS, D_INNER);
    }
    // selective scan
    scan_kernel<<<BB * 2, 256, 0, stream>>>(xdbl32, dtraw, xd32, A_log, dt_b, yv, l);
    // gated output activation -> f16
    yact_kernel<<<(MROWS * D_INNER) / 256, 256, 0, stream>>>(yv, xd32, xz, Dp, yact16, l);
    // delta = y_act @ out_w^T    [8192,256]  K=512, NT=4
    {
      int waves = (MROWS >> 4) * ((D_MODEL >> 4) / 4);
      gemm_wmma_kernel<D_INNER, 4><<<(waves + 7) / 8, 256, 0, stream>>>(
          yact16, D_INNER, outw16 + (size_t)l * D_MODEL * D_INNER, D_INNER,
          delta, D_MODEL, MROWS, D_MODEL);
    }
    // h = LayerNorm(h + delta)
    resid_ln_kernel<<<MROWS, D_MODEL, 0, stream>>>(delta, ln_g, ln_b, h32, h16);
  }

  mean_kernel<<<BB, D_MODEL, 0, stream>>>(h32, hmean);
  head_kernel<<<1, 128, 0, stream>>>(hmean, op_w, op_b, out);
}
